// MultiHeadCrossAttentionBlock_51281909514459
// MI455X (gfx1250) — matrix operand order repaired and checked by reading, not compiled, not run on hardware
//
#include <hip/hip_runtime.h>

typedef __attribute__((ext_vector_type(16))) __bf16 v16bf;
typedef __attribute__((ext_vector_type(8)))  __bf16 v8bf;
typedef __attribute__((ext_vector_type(8)))  float  v8f;
typedef __attribute__((ext_vector_type(8)))  unsigned short v8us;

#define HEADS 8
#define C 64
#define N 1024
#define CH 512
#define BATCH 8

// ---------- bf16 helpers ----------
__device__ __forceinline__ unsigned short f32_to_bf16_bits(float f) {
  unsigned int u = __builtin_bit_cast(unsigned int, f);
  unsigned int r = u + 0x7FFFu + ((u >> 16) & 1u);  // round-to-nearest-even
  return (unsigned short)(r >> 16);
}

// ---------- CDNA5 WMMA operand loaders (wave32 layouts per ISA 7.12.2) ----------
// A: 16x32 (MxK) bf16 from row-major [M x ld] source, K contiguous. Two 16B loads.
__device__ __forceinline__ v16bf load_a_tile(const unsigned short* __restrict__ src,
                                             int ld, int r0, int k0) {
  int lane = threadIdx.x & 31;
  int m  = lane & 15;
  int kb = (lane >> 4) << 3;  // 0 or 8
  const unsigned short* p = src + (size_t)(r0 + m) * ld + k0 + kb;
  union { v16bf v; v8bf h[2]; } u;
  u.h[0] = *(const v8bf*)(p);        // K = kb+0 .. kb+7
  u.h[1] = *(const v8bf*)(p + 16);   // K = kb+16 .. kb+23
  return u.v;
}
// B: 32x16 (KxN) bf16 where the SOURCE stores K contiguously:
// B[k][n] = src[(n0+n)*ld + k0 + k].  Two 16B loads per lane (same K order as A).
__device__ __forceinline__ v16bf load_bt_tile(const unsigned short* __restrict__ src,
                                              int ld, int k0, int n0) {
  int lane = threadIdx.x & 31;
  int nn = lane & 15;
  int kb = (lane >> 4) << 3;  // 0 or 8
  const unsigned short* p = src + (size_t)(n0 + nn) * ld + k0 + kb;
  union { v16bf v; v8bf h[2]; } u;
  u.h[0] = *(const v8bf*)(p);
  u.h[1] = *(const v8bf*)(p + 16);
  return u.v;
}
__device__ __forceinline__ v8f wmma_bf16(v16bf a, v16bf b, v8f c) {
  return __builtin_amdgcn_wmma_f32_16x16x32_bf16(false, a, false, b, (short)0, c,
                                                 false, false);
}

// ---------- 0) f32 -> bf16 weight conversion ----------
__global__ void convert_bf16_kernel(const float* __restrict__ src,
                                    unsigned short* __restrict__ dst, int n) {
  int i = blockIdx.x * blockDim.x + threadIdx.x;
  if (i < n) dst[i] = f32_to_bf16_bits(src[i]);
}

// ---------- 1) GroupNorm (32 groups of 2 channels x 1024) -> bf16, TRANSPOSED [b][n][c] ----------
__global__ void groupnorm_kernel(const float* __restrict__ xq, const float* __restrict__ xkv,
                                 const float* __restrict__ gq, const float* __restrict__ bq,
                                 const float* __restrict__ gkv, const float* __restrict__ bkv,
                                 unsigned short* __restrict__ xqnT,
                                 unsigned short* __restrict__ xkvnT) {
  __shared__ float rs[256], rq[256];
  int id = blockIdx.x;
  int tsel = id >> 8;        // 0 = x_q, 1 = x_kv
  int rem = id & 255;
  int b = rem >> 5, g = rem & 31;
  const float* x    = tsel ? xkv : xq;
  const float* gam  = tsel ? gkv : gq;
  const float* bet  = tsel ? bkv : bq;
  unsigned short* o = (tsel ? xkvnT : xqnT) + (size_t)b * N * C;
  const float* base = x + (size_t)(b * C + g * 2) * N;  // 2048 contiguous floats
  int tid = threadIdx.x;
  float s = 0.f, ss = 0.f;
  for (int i = tid; i < 2048; i += 256) { float v = base[i]; s += v; ss += v * v; }
  rs[tid] = s; rq[tid] = ss;
  __syncthreads();
  for (int off = 128; off > 0; off >>= 1) {
    if (tid < off) { rs[tid] += rs[tid + off]; rq[tid] += rq[tid + off]; }
    __syncthreads();
  }
  float mu  = rs[0] * (1.f / 2048.f);
  float var = rq[0] * (1.f / 2048.f) - mu * mu;
  float inv = rsqrtf(var + 1e-6f);
  for (int i = tid; i < 2048; i += 256) {
    int ch = g * 2 + (i >> 10);
    int n  = i & 1023;
    float v = (base[i] - mu) * inv * gam[ch] + bet[ch];
    o[(size_t)n * C + ch] = f32_to_bf16_bits(v);  // transposed: c contiguous
  }
}

// ---------- 2) Projection [512x64]x[64x1024] fused with spatial softmax ----------
// Block = 256 threads (8 waves), 16x1024 stripe in LDS. B operands double-buffered.
__global__ void proj_kernel(const unsigned short* __restrict__ W,
                            const float* __restrict__ bias,
                            const unsigned short* __restrict__ Xt,
                            unsigned short* __restrict__ Out,
                            int do_softmax, int store_transposed) {
  extern __shared__ float smem[];
  float* tile = smem;             // 16 * 1024
  float* redm = smem + 16 * N;    // 256
  float* reds = redm + 256;       // 256
  int b  = blockIdx.x >> 5;
  int r0 = (blockIdx.x & 31) * 16;
  const unsigned short* Xb = Xt + (size_t)b * N * C;
  int wave = threadIdx.x >> 5;
  int lane = threadIdx.x & 31;
  int nn = lane & 15, rb = (lane >> 4) << 3;
  // A reused across all 8 n-tiles of this wave
  v16bf a0 = load_a_tile(W, C, r0, 0);
  v16bf a1 = load_a_tile(W, C, r0, 32);
  int n0 = wave * 128;
  v16bf bc0 = load_bt_tile(Xb, C, 0,  n0);
  v16bf bc1 = load_bt_tile(Xb, C, 32, n0);
#pragma unroll
  for (int t = 0; t < 8; ++t) {
    v16bf bn0 = bc0, bn1 = bc1;
    if (t < 7) {  // prefetch next tile's B before consuming current
      bn0 = load_bt_tile(Xb, C, 0,  n0 + (t + 1) * 16);
      bn1 = load_bt_tile(Xb, C, 32, n0 + (t + 1) * 16);
    }
    v8f acc = {};
    acc = wmma_bf16(a0, bc0, acc);
    acc = wmma_bf16(a1, bc1, acc);
    int nt = n0 + t * 16;
#pragma unroll
    for (int r = 0; r < 8; ++r) tile[(rb + r) * N + nt + nn] = acc[r];
    bc0 = bn0; bc1 = bn1;
  }
  __syncthreads();
  int row = threadIdx.x & 15, seg = threadIdx.x >> 4;  // 16 threads per row
  float* trow = tile + row * N + seg * 64;
  if (do_softmax) {
    float mx = -3.4e38f;
    for (int j = 0; j < 64; ++j) mx = fmaxf(mx, trow[j]);
    redm[row * 16 + seg] = mx;
    __syncthreads();
    float m = redm[row * 16];
    for (int s2 = 1; s2 < 16; ++s2) m = fmaxf(m, redm[row * 16 + s2]);
    float sum = 0.f;
    for (int j = 0; j < 64; ++j) { float e = __expf(trow[j] - m); trow[j] = e; sum += e; }
    reds[row * 16 + seg] = sum;
    __syncthreads();
    float st = 0.f;
    for (int s2 = 0; s2 < 16; ++s2) st += reds[row * 16 + s2];
    float invs = 1.f / st;
    if (store_transposed) {
      unsigned short* Ob = Out + (size_t)b * N * CH + r0 + row;
      for (int j = 0; j < 64; ++j)
        Ob[(size_t)(seg * 64 + j) * CH] = f32_to_bf16_bits(trow[j] * invs);
    } else {
      unsigned short* Ob = Out + (size_t)(b * CH + r0 + row) * N + seg * 64;
      for (int j = 0; j < 64; ++j) Ob[j] = f32_to_bf16_bits(trow[j] * invs);
    }
  } else {
    float bb = bias[r0 + row];
    unsigned short* Ob = Out + (size_t)(b * CH + r0 + row) * N + seg * 64;
    for (int j = 0; j < 64; ++j) Ob[j] = f32_to_bf16_bits(trow[j] + bb);
  }
}

// ---------- 3) M = (V . K^T) / sqrt(c): per (b,h) 64x64, each wave a 32x32 block ----------
// 4 accumulators: 2 A-loads + 2 B-loads -> 4 WMMAs per k-step (k = n = 1024).
__global__ void vkT_kernel(const unsigned short* __restrict__ Vbf,
                           const unsigned short* __restrict__ Ksm,
                           unsigned short* __restrict__ Mbf) {
  int job = blockIdx.x * (blockDim.x >> 5) + (threadIdx.x >> 5);
  int bh = job >> 2, q = job & 3;                 // 2x2 grid of 32x32 blocks
  int r0 = (q >> 1) * 32, c0 = (q & 1) * 32;
  const unsigned short* Vp = Vbf + (size_t)bh * C * N;
  const unsigned short* Kp = Ksm + (size_t)bh * C * N;
  v8f acc00 = {}, acc01 = {}, acc10 = {}, acc11 = {};
  for (int k0 = 0; k0 < N; k0 += 32) {
    v16bf av0 = load_a_tile(Vp, N, r0, k0);
    v16bf av1 = load_a_tile(Vp, N, r0 + 16, k0);
    v16bf bk0 = load_bt_tile(Kp, N, k0, c0);
    v16bf bk1 = load_bt_tile(Kp, N, k0, c0 + 16);
    acc00 = wmma_bf16(av0, bk0, acc00);
    acc01 = wmma_bf16(av0, bk1, acc01);
    acc10 = wmma_bf16(av1, bk0, acc10);
    acc11 = wmma_bf16(av1, bk1, acc11);
  }
  int lane = threadIdx.x & 31, nn = lane & 15, rb = (lane >> 4) << 3;
  unsigned short* Mp = Mbf + (size_t)bh * C * C;
  auto store_m = [&](v8f acc, int rr, int cc) {
#pragma unroll
    for (int r = 0; r < 8; ++r)
      Mp[(rr + rb + r) * C + cc + nn] = f32_to_bf16_bits(acc[r] * 0.125f);
  };
  store_m(acc00, r0, c0);       store_m(acc01, r0, c0 + 16);
  store_m(acc10, r0 + 16, c0);  store_m(acc11, r0 + 16, c0 + 16);
}

// ---------- 4) out = M . Q per (b,h), each wave a 32x32 block -> [b][n][512] packed ----------
__global__ void mq_kernel(const unsigned short* __restrict__ Mbf,
                          const unsigned short* __restrict__ QsmT,
                          unsigned short* __restrict__ ObfT) {
  int job = blockIdx.x * (blockDim.x >> 5) + (threadIdx.x >> 5);
  int bh = job >> 6, sub = job & 63;              // 2 x 32 blocks of 32x32
  int b = bh >> 3, h = bh & 7;
  int r0 = (sub >> 5) * 32, n0 = (sub & 31) * 32;
  const unsigned short* Mp = Mbf + (size_t)bh * C * C;
  const unsigned short* Qb = QsmT + (size_t)b * N * CH;   // [n][512], c contiguous
  v8f acc00 = {}, acc01 = {}, acc10 = {}, acc11 = {};
#pragma unroll
  for (int k0 = 0; k0 < C; k0 += 32) {
    v16bf a0 = load_a_tile(Mp, C, r0, k0);
    v16bf a1 = load_a_tile(Mp, C, r0 + 16, k0);
    v16bf b0 = load_bt_tile(Qb, CH, h * C + k0, n0);
    v16bf b1 = load_bt_tile(Qb, CH, h * C + k0, n0 + 16);
    acc00 = wmma_bf16(a0, b0, acc00);
    acc01 = wmma_bf16(a0, b1, acc01);
    acc10 = wmma_bf16(a1, b0, acc10);
    acc11 = wmma_bf16(a1, b1, acc11);
  }
  int lane = threadIdx.x & 31, nn = lane & 15, rb = (lane >> 4) << 3;
  auto store_t = [&](v8f acc, int rr, int cc) {  // one packed 16B store per lane
    v8us pk;
#pragma unroll
    for (int r = 0; r < 8; ++r) pk[r] = f32_to_bf16_bits(acc[r]);
    *(v8us*)(ObfT + (size_t)(b * N + cc + nn) * CH + h * C + rr + rb) = pk;
  };
  store_t(acc00, r0, n0);       store_t(acc01, r0, n0 + 16);
  store_t(acc10, r0 + 16, n0);  store_t(acc11, r0 + 16, n0 + 16);
}

// ---------- 5) Y = Wo . out + bo per batch, each wave a 32x32 block -> f32 ----------
__global__ void outproj_kernel(const unsigned short* __restrict__ Wobf,
                               const float* __restrict__ bo,
                               const unsigned short* __restrict__ ObfT,
                               float* __restrict__ Y) {
  int job = blockIdx.x * (blockDim.x >> 5) + (threadIdx.x >> 5);
  int b = job >> 6, sub = job & 63;               // 2 x 32 blocks of 32x32
  int r0 = (sub >> 5) * 32, n0 = (sub & 31) * 32;
  const unsigned short* Bp = ObfT + (size_t)b * N * CH;   // [n][512], ch contiguous
  v8f acc00 = {}, acc01 = {}, acc10 = {}, acc11 = {};
  for (int k0 = 0; k0 < CH; k0 += 32) {
    v16bf a0 = load_a_tile(Wobf, CH, r0, k0);
    v16bf a1 = load_a_tile(Wobf, CH, r0 + 16, k0);
    v16bf b0 = load_bt_tile(Bp, CH, k0, n0);
    v16bf b1 = load_bt_tile(Bp, CH, k0, n0 + 16);
    acc00 = wmma_bf16(a0, b0, acc00);
    acc01 = wmma_bf16(a0, b1, acc01);
    acc10 = wmma_bf16(a1, b0, acc10);
    acc11 = wmma_bf16(a1, b1, acc11);
  }
  int lane = threadIdx.x & 31, nn = lane & 15, rb = (lane >> 4) << 3;
  float* Yp = Y + (size_t)b * C * N;
  auto store_y = [&](v8f acc, int rr, int cc) {
#pragma unroll
    for (int r = 0; r < 8; ++r)
      Yp[(rr + rb + r) * N + cc + nn] = acc[r] + bo[rr + rb + r];
  };
  store_y(acc00, r0, n0);       store_y(acc01, r0, n0 + 16);
  store_y(acc10, r0 + 16, n0);  store_y(acc11, r0 + 16, n0 + 16);
}

extern "C" void kernel_launch(void* const* d_in, const int* in_sizes, int n_in,
                              void* d_out, int out_size, void* d_ws, size_t ws_size,
                              hipStream_t stream) {
  (void)in_sizes; (void)n_in; (void)out_size; (void)ws_size;
  const float* x_q  = (const float*)d_in[0];
  const float* x_kv = (const float*)d_in[1];
  const float* gq   = (const float*)d_in[2];
  const float* btq  = (const float*)d_in[3];
  const float* gkv  = (const float*)d_in[4];
  const float* btkv = (const float*)d_in[5];
  const float* Wq   = (const float*)d_in[6];
  // bq (d_in[7]) / bk (d_in[9]) dropped: softmax over n is shift-invariant.
  const float* Wk   = (const float*)d_in[8];
  const float* Wv   = (const float*)d_in[10];
  const float* bv   = (const float*)d_in[11];
  const float* Wo   = (const float*)d_in[12];
  const float* bo   = (const float*)d_in[13];
  float* Y = (float*)d_out;

  unsigned short* ws = (unsigned short*)d_ws;
  size_t off = 0;
  unsigned short* Wq_bf = ws + off; off += (size_t)CH * C;
  unsigned short* Wk_bf = ws + off; off += (size_t)CH * C;
  unsigned short* Wv_bf = ws + off; off += (size_t)CH * C;
  unsigned short* Wo_bf = ws + off; off += (size_t)C * CH;
  unsigned short* xqnT  = ws + off; off += (size_t)BATCH * N * C;   // [b][n][c]
  unsigned short* xkvnT = ws + off; off += (size_t)BATCH * N * C;   // [b][n][c]
  unsigned short* QsmT  = ws + off; off += (size_t)BATCH * N * CH;  // [b][n][512]
  unsigned short* Ksm   = ws + off; off += (size_t)BATCH * CH * N;  // [b][512][n]
  unsigned short* Vbf   = ws + off; off += (size_t)BATCH * CH * N;  // [b][512][n]
  unsigned short* Mbf   = ws + off; off += (size_t)BATCH * HEADS * C * C;
  unsigned short* ObfT  = ws + off; off += (size_t)BATCH * N * CH;  // [b][n][512]

  convert_bf16_kernel<<<(CH * C + 255) / 256, 256, 0, stream>>>(Wq, Wq_bf, CH * C);
  convert_bf16_kernel<<<(CH * C + 255) / 256, 256, 0, stream>>>(Wk, Wk_bf, CH * C);
  convert_bf16_kernel<<<(CH * C + 255) / 256, 256, 0, stream>>>(Wv, Wv_bf, CH * C);
  convert_bf16_kernel<<<(C * CH + 255) / 256, 256, 0, stream>>>(Wo, Wo_bf, C * CH);

  groupnorm_kernel<<<512, 256, 0, stream>>>(x_q, x_kv, gq, btq, gkv, btkv, xqnT, xkvnT);

  size_t projLds = (size_t)(16 * N + 512) * sizeof(float);  // ~66 KB (320 KB WGP LDS)
  proj_kernel<<<BATCH * 32, 256, projLds, stream>>>(Wq_bf, bo, xqnT,  QsmT, 1, 1);
  proj_kernel<<<BATCH * 32, 256, projLds, stream>>>(Wk_bf, bo, xkvnT, Ksm,  1, 0);
  proj_kernel<<<BATCH * 32, 256, projLds, stream>>>(Wv_bf, bv, xkvnT, Vbf,  0, 0);

  vkT_kernel<<<32, 256, 0, stream>>>(Vbf, Ksm, Mbf);          // 256 wave-jobs (32x32 each)
  mq_kernel<<<512, 256, 0, stream>>>(Mbf, QsmT, ObfT);        // 4096 wave-jobs
  outproj_kernel<<<64, 256, 0, stream>>>(Wo_bf, bo, ObfT, Y); // 512 wave-jobs
}
